// Decoder_31370441129997
// MI455X (gfx1250) — matrix operand order repaired and checked by reading, not compiled, not run on hardware
//
#include <hip/hip_runtime.h>
#include <hip/hip_bf16.h>

// ---------------------------------------------------------------------------
// GRU decoder rollout (TPRED=12 steps) on MI455X (gfx1250, wave32, WMMA).
// All GEMMs run through v_wmma_f32_16x16x32_bf16 with fp32 accumulation.
// Block structure: 8 waves share double-buffered LDS-staged 32-row A slabs
// (same batch rows); each wave covers a distinct 16-wide hid column tile and
// computes a 32x16 output tile (2 M-subtiles -> 2x weight reuse, 12-way WMMA
// ILP in the fused GRU kernel). One barrier per K-step: stage slab kb+32
// while computing kb.
// ---------------------------------------------------------------------------

#define B_SZ   4096
#define ENC    512
#define HID    512
#define NLAYER 3
#define TPRED  12

#define LDS_STRIDE 40   // 32 + 8 pad (80B/row: 16B-aligned, 20-bank shift)
#define SLAB_ELEMS (32 * LDS_STRIDE)

typedef __attribute__((ext_vector_type(16))) __bf16 v16bf;
typedef __attribute__((ext_vector_type(8)))  __bf16 v8bf;
typedef __attribute__((ext_vector_type(4)))  __bf16 v4bf;
typedef __attribute__((ext_vector_type(8)))  float  v8f;

// ---- WMMA fragment loaders (contiguous k order, same in A and B) ------------------
// A matrix 16x32 bf16 from an LDS slab (rows x LDS_STRIDE):
//   lane L: M = row0 + (L&15); khalf = (L>>4)*16
//   elements 0..7  -> K = khalf + {0..7}
//   elements 8..15 -> K = khalf + 8 + {0..7}
__device__ __forceinline__ v16bf lds_a_frag(const __bf16* S, int row0, int lane) {
    const __bf16* row = S + (row0 + (lane & 15)) * LDS_STRIDE;
    const int kh = (lane >> 4) * 16;
    v8bf c0 = *(const v8bf*)(row + kh);        // K = kh..kh+7        (16B)
    v8bf c1 = *(const v8bf*)(row + 8 + kh);    // K = 8+kh..8+kh+7    (16B)
    v16bf a;
#pragma unroll
    for (int i = 0; i < 8; ++i) { a[i] = c0[i]; a[i + 8] = c1[i]; }
    return a;
}

// B matrix 32x16 bf16 where B[k][n] = W[n0+n][k] (row-major W[N][512]):
//   lane L: N = n0 + (L&15); K = kb + (L>>4)*16 + {0..15} contiguous -> one
//   aligned 32-byte load per lane.
__device__ __forceinline__ v16bf load_b_frag(const __bf16* __restrict__ W,
                                             int n0, int kb, int lane) {
    const int n = n0 + (lane & 15);
    const int k = kb + (lane >> 4) * 16;
    return *(const v16bf*)(W + (size_t)n * 512 + k);
}

__device__ __forceinline__ float sigmoidf_(float x) {
    return 1.0f / (1.0f + __expf(-x));
}

__device__ __forceinline__ v8f wmma_bf16(v16bf a, v16bf b, v8f c) {
    return __builtin_amdgcn_wmma_f32_16x16x32_bf16(
        false, a, false, b, (short)0, c, false, false);
}

// ---- x = relu(A @ W^T + b), A:[4096][512] bf16, W:[512][512] bf16 ----------
// Grid: 512 blocks = (4096/32 M-blocks) x (4 N-groups), 8 waves/block.
__global__ void __launch_bounds__(256)
gemm_relu_kernel(const __bf16* __restrict__ A, const __bf16* __restrict__ W,
                 const float* __restrict__ bias, __bf16* __restrict__ X) {
    __shared__ __bf16 As[2][SLAB_ELEMS];

    const int lane = threadIdx.x & 31;
    const int wave = threadIdx.x >> 5;
    const int bm = blockIdx.x >> 2;              // 0..127 -> m0 = bm*32
    const int bn = blockIdx.x & 3;               // 0..3
    const int m0 = bm << 5;
    const int n0 = (bn << 7) + (wave << 4);      // 8 waves cover 128 cols

    // cooperative A-slab staging: 256 threads x 4 bf16 (8B) = 32x32
    const int lr = threadIdx.x >> 3;             // 0..31
    const int lc = (threadIdx.x & 7) << 2;       // 0,4,...,28
    const __bf16* gsrc = A + (size_t)(m0 + lr) * 512 + lc;

    // preload slab for kb = 0
    *(v4bf*)(As[0] + lr * LDS_STRIDE + lc) = *(const v4bf*)(gsrc);

    v8f acc[2] = {{}, {}};

    for (int kb = 0; kb < 512; kb += 32) {
        const int p = (kb >> 5) & 1;
        __syncthreads();                          // slab p ready, slab p^1 free
        if (kb + 32 < 512) {
            *(v4bf*)(As[p ^ 1] + lr * LDS_STRIDE + lc) =
                *(const v4bf*)(gsrc + kb + 32);
            __builtin_prefetch(W + (size_t)(n0 + (lane & 15)) * 512 + kb + 32, 0, 3);
        }
        v16bf b = load_b_frag(W, n0, kb, lane);
#pragma unroll
        for (int s = 0; s < 2; ++s) {
            v16bf a = lds_a_frag(As[p], s * 16, lane);
            acc[s] = wmma_bf16(a, b, acc[s]);
        }
    }

    const int nc = n0 + (lane & 15);
    const float bn_ = bias[nc];
#pragma unroll
    for (int s = 0; s < 2; ++s) {
#pragma unroll
        for (int r = 0; r < 8; ++r) {
            const int m = m0 + s * 16 + (lane >> 4) * 8 + r;
            float v = acc[s][r] + bn_;
            v = v > 0.0f ? v : 0.0f;
            X[(size_t)m * 512 + nc] = (__bf16)v;
        }
    }
}

// ---- fused GRU cell: h' = (1-z)*n + z*h ------------------------------------
// Each wave computes a 32x16 (batch x hid) tile: 12 accumulators
// (2 M-subtiles x {r,z,n} gates x {gi, gh}).  X/H slabs double-buffered in
// LDS and shared by all 8 waves of the block.
__global__ void __launch_bounds__(256)
gru_kernel(const __bf16* __restrict__ X, const __bf16* __restrict__ Hin,
           __bf16* __restrict__ Hout,
           const __bf16* __restrict__ Wi, const __bf16* __restrict__ Wh,
           const float* __restrict__ bi, const float* __restrict__ bh) {
    __shared__ __bf16 Xs[2][SLAB_ELEMS];
    __shared__ __bf16 Hs[2][SLAB_ELEMS];

    const int lane = threadIdx.x & 31;
    const int wave = threadIdx.x >> 5;
    const int bm = blockIdx.x >> 2;              // 0..127
    const int bn = blockIdx.x & 3;               // 0..3
    const int m0 = bm << 5;
    const int n0 = (bn << 7) + (wave << 4);

    // cooperative staging: 128 threads per matrix, 8 bf16 (16B) each
    const int half = threadIdx.x >> 7;           // 0 -> Xs, 1 -> Hs
    const int lr = (threadIdx.x & 127) >> 2;     // 0..31
    const int lc = (threadIdx.x & 3) << 3;       // 0,8,16,24
    const __bf16* gsrc =
        (half ? Hin : X) + (size_t)(m0 + lr) * 512 + lc;
    __bf16* lds0 = (half ? Hs[0] : Xs[0]) + lr * LDS_STRIDE + lc;
    __bf16* lds1 = (half ? Hs[1] : Xs[1]) + lr * LDS_STRIDE + lc;

    // preload slab for kb = 0
    *(v8bf*)lds0 = *(const v8bf*)(gsrc);

    v8f gi[2][3] = {{{}, {}, {}}, {{}, {}, {}}};
    v8f gh[2][3] = {{{}, {}, {}}, {{}, {}, {}}};

    for (int kb = 0; kb < 512; kb += 32) {
        const int p = (kb >> 5) & 1;
        __syncthreads();                          // slab p ready, slab p^1 free
        if (kb + 32 < 512) {
            *(v8bf*)(p ? lds0 : lds1) = *(const v8bf*)(gsrc + kb + 32);
            __builtin_prefetch(Wi + (size_t)(n0 + (lane & 15)) * 512 + kb + 32, 0, 3);
            __builtin_prefetch(Wh + (size_t)(n0 + (lane & 15)) * 512 + kb + 32, 0, 3);
        }

        v16bf ax[2], ah[2];
#pragma unroll
        for (int s = 0; s < 2; ++s) {
            ax[s] = lds_a_frag(Xs[p], s * 16, lane);
            ah[s] = lds_a_frag(Hs[p], s * 16, lane);
        }
#pragma unroll
        for (int g = 0; g < 3; ++g) {
            v16bf bwi = load_b_frag(Wi, n0 + g * 512, kb, lane);
            v16bf bwh = load_b_frag(Wh, n0 + g * 512, kb, lane);
#pragma unroll
            for (int s = 0; s < 2; ++s) {
                gi[s][g] = wmma_bf16(ax[s], bwi, gi[s][g]);
                gh[s][g] = wmma_bf16(ah[s], bwh, gh[s][g]);
            }
        }
    }

    const int nc = n0 + (lane & 15);
    const float bir = bi[nc], biz = bi[nc + 512], bin_ = bi[nc + 1024];
    const float bhr = bh[nc], bhz = bh[nc + 512], bhn = bh[nc + 1024];

#pragma unroll
    for (int s = 0; s < 2; ++s) {
#pragma unroll
        for (int r = 0; r < 8; ++r) {
            const int m = m0 + s * 16 + (lane >> 4) * 8 + r;
            const float hv = (float)Hin[(size_t)m * 512 + nc];
            const float rg = sigmoidf_(gi[s][0][r] + bir + gh[s][0][r] + bhr);
            const float zg = sigmoidf_(gi[s][1][r] + biz + gh[s][1][r] + bhz);
            const float ng = tanhf(gi[s][2][r] + bin_ + rg * (gh[s][2][r] + bhn));
            const float hn = (1.0f - zg) * ng + zg * hv;
            Hout[(size_t)m * 512 + nc] = (__bf16)hn;
        }
    }
}

// ---- out[b, t, :] = h2[b] @ Wout^T + bout (tiny N=2 projection) ------------
__global__ void __launch_bounds__(256)
out_kernel(const __bf16* __restrict__ H2, const float* __restrict__ Wout,
           const float* __restrict__ bout, float* __restrict__ out, int t) {
    const int idx = blockIdx.x * blockDim.x + threadIdx.x;
    if (idx >= B_SZ * 2) return;
    const int b = idx >> 1, o = idx & 1;
    const __bf16* hrow = H2 + (size_t)b * 512;
    const float* wrow = Wout + o * 512;
    float s = 0.0f;
#pragma unroll 8
    for (int k = 0; k < 512; ++k) s += (float)hrow[k] * wrow[k];
    out[(size_t)b * (TPRED * 2) + t * 2 + o] = s + bout[o];
}

// ---- fp32 -> bf16 conversion / zero-init -----------------------------------
__global__ void cvt_kernel(const float* __restrict__ src,
                           __bf16* __restrict__ dst, int n) {
    for (int i = blockIdx.x * blockDim.x + threadIdx.x; i < n;
         i += gridDim.x * blockDim.x)
        dst[i] = (__bf16)src[i];
}

__global__ void zero_kernel(__bf16* __restrict__ dst, int n) {
    for (int i = blockIdx.x * blockDim.x + threadIdx.x; i < n;
         i += gridDim.x * blockDim.x)
        dst[i] = (__bf16)0.0f;
}

// ---------------------------------------------------------------------------
extern "C" void kernel_launch(void* const* d_in, const int* in_sizes, int n_in,
                              void* d_out, int out_size, void* d_ws, size_t ws_size,
                              hipStream_t stream) {
    const float* rep   = (const float*)d_in[0];   // (4096, 1, 512)
    const float* W_in  = (const float*)d_in[2];   // (512, 512)
    const float* b_in  = (const float*)d_in[3];   // (512,)
    const float* W_ih  = (const float*)d_in[4];   // (3, 1536, 512)
    const float* W_hh  = (const float*)d_in[5];   // (3, 1536, 512)
    const float* b_ih  = (const float*)d_in[6];   // (3, 1536)
    const float* b_hh  = (const float*)d_in[7];   // (3, 1536)
    const float* W_out = (const float*)d_in[8];   // (2, 512)
    const float* b_out = (const float*)d_in[9];   // (2,)
    float* out = (float*)d_out;                   // (4096, 12, 2)

    // ---- workspace carve-out (~42 MB) ----
    char* ws = (char*)d_ws;
    size_t off = 0;
    auto carve = [&](size_t bytes) -> void* {
        void* p = ws + off;
        off = (off + bytes + 255) & ~(size_t)255;
        return p;
    };
    const size_t W_IN_N = (size_t)512 * 512;
    const size_t W_G_N  = (size_t)NLAYER * 1536 * 512;
    const size_t ACT_N  = (size_t)B_SZ * 512;

    __bf16* Winb = (__bf16*)carve(W_IN_N * 2);
    __bf16* Wib  = (__bf16*)carve(W_G_N * 2);
    __bf16* Whb  = (__bf16*)carve(W_G_N * 2);
    __bf16* repb = (__bf16*)carve(ACT_N * 2);
    __bf16* xb   = (__bf16*)carve(ACT_N * 2);
    __bf16* hbuf = (__bf16*)carve((size_t)2 * NLAYER * ACT_N * 2); // ping/pong

    __bf16* h[2][NLAYER];
    for (int p = 0; p < 2; ++p)
        for (int l = 0; l < NLAYER; ++l)
            h[p][l] = hbuf + ((size_t)p * NLAYER + l) * ACT_N;

    // ---- one-time conversions + state init (re-done each launch: deterministic)
    cvt_kernel<<<512, 256, 0, stream>>>(W_in, Winb, (int)W_IN_N);
    cvt_kernel<<<2048, 256, 0, stream>>>(W_ih, Wib, (int)W_G_N);
    cvt_kernel<<<2048, 256, 0, stream>>>(W_hh, Whb, (int)W_G_N);
    cvt_kernel<<<2048, 256, 0, stream>>>(rep, repb, (int)ACT_N);
    zero_kernel<<<2048, 256, 0, stream>>>(h[0][0], (int)(NLAYER * ACT_N));

    // ---- rollout ----
    const __bf16* rptr = repb;
    int cur = 0;
    for (int t = 0; t < TPRED; ++t) {
        // x = relu(rep @ W_in^T + b_in)
        gemm_relu_kernel<<<512, 256, 0, stream>>>(rptr, Winb, b_in, xb);

        const __bf16* inp = xb;
        for (int l = 0; l < NLAYER; ++l) {
            gru_kernel<<<512, 256, 0, stream>>>(
                inp, h[cur][l], h[1 - cur][l],
                Wib + (size_t)l * 1536 * 512, Whb + (size_t)l * 1536 * 512,
                b_ih + (size_t)l * 1536, b_hh + (size_t)l * 1536);
            inp = h[1 - cur][l];
        }

        out_kernel<<<(B_SZ * 2 + 255) / 256, 256, 0, stream>>>(
            h[1 - cur][NLAYER - 1], W_out, b_out, out, t);

        rptr = h[1 - cur][NLAYER - 1];
        cur ^= 1;
    }
}
